// TransformerBlock_4715874091819
// MI455X (gfx1250) — matrix operand order repaired and checked by reading, not compiled, not run on hardware
//
#include <hip/hip_runtime.h>

// ---------------------------------------------------------------------------
// CDNA5 / gfx1250 transformer block. wave32. All GEMMs via
// v_wmma_f32_16x16x32_bf16 (fp32 accumulate). Compile-only environment.
// GEMM: 32x64 output tile per wave, software-pipelined k-loop (distinct
// current/next register tiles so loads overlap WMMAs instead of
// s_wait_loadcnt 0x0 serialization).
// ---------------------------------------------------------------------------

typedef __bf16 bf16;
typedef __attribute__((ext_vector_type(8)))  __bf16 bf16x8;
typedef __attribute__((ext_vector_type(16))) __bf16 bf16x16;
typedef __attribute__((ext_vector_type(8)))  float  floatx8;

union AV { bf16x16 v; struct { bf16x8 lo; bf16x8 hi; } s; };

#define B_  4
#define T_  2048
#define E_  1024
#define H_  16
#define D_  64
#define FF_ 4096
#define BT_ (B_ * T_)

static __device__ __forceinline__ floatx8 wmma_bf16(bf16x16 a, bf16x16 b, floatx8 c) {
  return __builtin_amdgcn_wmma_f32_16x16x32_bf16(
      /*neg_a=*/false, a, /*neg_b=*/false, b,
      /*c_mod=*/(short)0, c, /*reuse_a=*/false, /*reuse_b=*/false);
}

// A tile 16x32 (lane = row, K = half*16 + 0..15 contiguous)
static __device__ __forceinline__ AV loadA(const bf16* Arow, int k0, int half) {
  AV a;
  a.s.lo = *(const bf16x8*)(Arow + k0 + half * 16);
  a.s.hi = *(const bf16x8*)(Arow + k0 + half * 16 + 8);
  return a;
}
// B tile 32x16 = W^T (lane = W row / output col, K = half*16 + 0..15 contiguous)
static __device__ __forceinline__ AV loadB(const bf16* Wrow, int k0, int half) {
  const bf16* p = Wrow + k0 + half * 16;
  AV b;
  b.s.lo = *(const bf16x8*)p;
  b.s.hi = *(const bf16x8*)(p + 8);
  return b;
}

// ---------------------------------------------------------------------------
// fp32 -> bf16 conversion (weights converted once)
// ---------------------------------------------------------------------------
__global__ __launch_bounds__(256) void cvt_bf16_kernel(const float* __restrict__ in,
                                                       bf16* __restrict__ out, int n) {
  int i = blockIdx.x * blockDim.x + threadIdx.x;
  int stride = gridDim.x * blockDim.x;
  for (; i < n; i += stride) out[i] = (bf16)in[i];
}

// ---------------------------------------------------------------------------
// LayerNorm: one block per row of E_=1024, 256 threads (8 waves), bf16 output
// ---------------------------------------------------------------------------
__global__ __launch_bounds__(256) void ln_kernel(const float* __restrict__ X,
                                                 const float* __restrict__ g,
                                                 const float* __restrict__ be,
                                                 bf16* __restrict__ out) {
  __shared__ float red[8];
  __shared__ float stat[2];
  const int row = blockIdx.x;
  const float* xr = X + (size_t)row * E_;

  float v[4];
  float s = 0.f;
#pragma unroll
  for (int i = 0; i < 4; ++i) { v[i] = xr[threadIdx.x * 4 + i]; s += v[i]; }
#pragma unroll
  for (int off = 16; off > 0; off >>= 1) s += __shfl_xor(s, off, 32);
  if ((threadIdx.x & 31) == 0) red[threadIdx.x >> 5] = s;
  __syncthreads();
  if (threadIdx.x == 0) {
    float t = 0.f;
#pragma unroll
    for (int i = 0; i < 8; ++i) t += red[i];
    stat[0] = t * (1.f / (float)E_);
  }
  __syncthreads();
  const float mean = stat[0];

  float q = 0.f;
#pragma unroll
  for (int i = 0; i < 4; ++i) { float d = v[i] - mean; q += d * d; }
#pragma unroll
  for (int off = 16; off > 0; off >>= 1) q += __shfl_xor(q, off, 32);
  __syncthreads();
  if ((threadIdx.x & 31) == 0) red[threadIdx.x >> 5] = q;
  __syncthreads();
  if (threadIdx.x == 0) {
    float t = 0.f;
#pragma unroll
    for (int i = 0; i < 8; ++i) t += red[i];
    stat[1] = rsqrtf(t * (1.f / (float)E_) + 1e-5f);
  }
  __syncthreads();
  const float rstd = stat[1];

#pragma unroll
  for (int i = 0; i < 4; ++i) {
    int c = threadIdx.x * 4 + i;
    out[(size_t)row * E_ + c] = (bf16)((v[i] - mean) * rstd * g[c] + be[c]);
  }
}

// ---------------------------------------------------------------------------
// WMMA GEMM: C[M,N] = A[M,K](bf16,row-major) * W[N,K]^T (torch Linear layout).
// One wave computes a 32x64 output tile: 8 f32 accumulators, per k-step
// 2 A-tiles + 4 B-tiles -> 8 WMMAs. k-loop software-pipelined: next-iteration
// tiles are loaded into separate registers before the current WMMAs issue.
// Fused epilogue: +bias, optional relu, optional fp32 residual, f32/bf16 out.
// ---------------------------------------------------------------------------
__global__ __launch_bounds__(128) void gemm_bf16_wmma(
    const bf16* __restrict__ A, const bf16* __restrict__ W,
    const float* __restrict__ bias, const float* __restrict__ resid,
    float* __restrict__ Cf, bf16* __restrict__ Cb,
    int M, int N, int K, int relu) {
  const int wave   = blockIdx.x * (blockDim.x >> 5) + (threadIdx.x >> 5);
  const int lane   = threadIdx.x & 31;
  const int half   = lane >> 4;
  const int lrow   = lane & 15;
  const int tilesN = N >> 6;
  const int mTile  = wave / tilesN;   // 32-row tiles
  const int nTile  = wave % tilesN;   // 64-col tiles
  if (mTile * 32 >= M) return;

  const bf16* A0  = A + (size_t)(mTile * 32 + lrow) * K;
  const bf16* A1  = A0 + (size_t)16 * K;
  const bf16* Wr0 = W + (size_t)(nTile * 64 + 0 * 16 + lrow) * K;
  const bf16* Wr1 = W + (size_t)(nTile * 64 + 1 * 16 + lrow) * K;
  const bf16* Wr2 = W + (size_t)(nTile * 64 + 2 * 16 + lrow) * K;
  const bf16* Wr3 = W + (size_t)(nTile * 64 + 3 * 16 + lrow) * K;

  floatx8 acc[2][4] = {};

  // Prologue: load k=0 tiles.
  AV a0 = loadA(A0, 0, half), a1 = loadA(A1, 0, half);
  AV b0 = loadB(Wr0, 0, half), b1 = loadB(Wr1, 0, half);
  AV b2 = loadB(Wr2, 0, half), b3 = loadB(Wr3, 0, half);

  for (int kn = 32; kn <= K; kn += 32) {
    // Capture current tiles, then issue next-iteration loads (stay in flight
    // while the 8 WMMAs below execute).
    AV a0c = a0, a1c = a1, b0c = b0, b1c = b1, b2c = b2, b3c = b3;
    if (kn < K) {
      a0 = loadA(A0, kn, half);
      a1 = loadA(A1, kn, half);
      b0 = loadB(Wr0, kn, half);
      b1 = loadB(Wr1, kn, half);
      b2 = loadB(Wr2, kn, half);
      b3 = loadB(Wr3, kn, half);
      __builtin_prefetch((const void*)(A0 + kn + 32), 0, 3);
      __builtin_prefetch((const void*)(Wr0 + kn + 32), 0, 3);
      __builtin_prefetch((const void*)(Wr2 + kn + 32), 0, 3);
    }
    acc[0][0] = wmma_bf16(a0c.v, b0c.v, acc[0][0]);
    acc[1][0] = wmma_bf16(a1c.v, b0c.v, acc[1][0]);
    acc[0][1] = wmma_bf16(a0c.v, b1c.v, acc[0][1]);
    acc[1][1] = wmma_bf16(a1c.v, b1c.v, acc[1][1]);
    acc[0][2] = wmma_bf16(a0c.v, b2c.v, acc[0][2]);
    acc[1][2] = wmma_bf16(a1c.v, b2c.v, acc[1][2]);
    acc[0][3] = wmma_bf16(a0c.v, b3c.v, acc[0][3]);
    acc[1][3] = wmma_bf16(a1c.v, b3c.v, acc[1][3]);
  }

#pragma unroll
  for (int mt = 0; mt < 2; ++mt) {
#pragma unroll
    for (int nt = 0; nt < 4; ++nt) {
      const int col = nTile * 64 + nt * 16 + lrow;
      const float bb = bias ? bias[col] : 0.f;
#pragma unroll
      for (int r = 0; r < 8; ++r) {
        const int row = mTile * 32 + mt * 16 + half * 8 + r;  // C layout
        float vv = acc[mt][nt][r] + bb;
        if (relu) vv = vv > 0.f ? vv : 0.f;
        if (resid) vv += resid[(size_t)row * N + col];
        if (Cf) Cf[(size_t)row * N + col] = vv;
        if (Cb) Cb[(size_t)row * N + col] = (bf16)vv;
      }
    }
  }
}

// ---------------------------------------------------------------------------
// Causal flash attention, one wave per 16-query-row tile of one (b,h).
// Streams keys 32 at a time: S = Q K^T (4 WMMAs), online softmax in registers,
// P restaged C-layout -> A-layout via per-wave LDS, O += P V (4 WMMAs, V
// staged in LDS, read back in B-layout). scale = 1/sqrt(E) per the reference.
// ---------------------------------------------------------------------------
__global__ __launch_bounds__(128) void attn_kernel(const bf16* __restrict__ Qb,
                                                   const bf16* __restrict__ Kb,
                                                   const bf16* __restrict__ Vb,
                                                   bf16* __restrict__ Ob) {
  __shared__ __align__(16) bf16 Pl[4][16 * 32];   // per-wave P tile
  __shared__ __align__(16) bf16 Vl[4][32 * 64];   // per-wave V tile

  const int w    = threadIdx.x >> 5;
  const int lane = threadIdx.x & 31;
  const int half = lane >> 4;
  const int lrow = lane & 15;

  const int tile  = blockIdx.x * 4 + w;           // B_*H_*(T_/16) tiles total
  const int b     = tile / (H_ * (T_ / 16));
  const int rem   = tile % (H_ * (T_ / 16));
  const int h     = rem / (T_ / 16);
  const int qBase = (rem % (T_ / 16)) * 16;

  const size_t headOff = (size_t)b * T_ * E_ + (size_t)h * D_;

  // Q tile 16x64 in A-layout, split into two K=32 chunks.
  AV qa[2];
  {
    const bf16* qrow = Qb + headOff + (size_t)(qBase + lrow) * E_;
#pragma unroll
    for (int c = 0; c < 2; ++c) qa[c] = loadA(qrow, c * 32, half);
  }

  floatx8 o[4] = {};
  float mrow[8], lsum[8];
#pragma unroll
  for (int r = 0; r < 8; ++r) { mrow[r] = -1e30f; lsum[r] = 0.f; }

  const float scale = 0.03125f;  // 1/sqrt(E_) == 1/32 (reference scales by embd)

  const int jmax = (qBase + 15) >> 5;
  for (int j = 0; j <= jmax; ++j) {
    const int kb = j * 32;

    // Stage V tile [32 keys x 64 d] into LDS (lane copies key row kb+lane).
    {
      const bf16x8* src = (const bf16x8*)(Vb + headOff + (size_t)(kb + lane) * E_);
      bf16x8* dst = (bf16x8*)&Vl[w][lane * 64];
#pragma unroll
      for (int i = 0; i < 8; ++i) dst[i] = src[i];
    }

    // S = Q K^T for two 16-key sub-tiles (contraction over d in two chunks).
    floatx8 s[2] = {};
#pragma unroll
    for (int st = 0; st < 2; ++st) {
      const bf16* krow = Kb + headOff + (size_t)(kb + st * 16 + lrow) * E_;
#pragma unroll
      for (int c = 0; c < 2; ++c) {
        AV bv;  // B 32x16: lane col = key, K = c*32 + half*16 + (0..15) along d
        bv.s.lo = *(const bf16x8*)(krow + c * 32 + half * 16);
        bv.s.hi = *(const bf16x8*)(krow + c * 32 + half * 16 + 8);
        s[st] = wmma_bf16(qa[c].v, bv.v, s[st]);
      }
    }

    // Scale + causal mask (C layout: row = qBase + 8*half + r, key = kb+st*16+lrow)
#pragma unroll
    for (int st = 0; st < 2; ++st) {
      const int key = kb + st * 16 + lrow;
#pragma unroll
      for (int r = 0; r < 8; ++r) {
        const int qr = qBase + half * 8 + r;
        float vv = s[st][r] * scale;
        if (key > qr) vv = -1e30f;
        s[st][r] = vv;
      }
    }

    // Online softmax: row reductions across the 16 lanes of each half-group.
#pragma unroll
    for (int r = 0; r < 8; ++r) {
      float mx = fmaxf(s[0][r], s[1][r]);
#pragma unroll
      for (int off = 1; off < 16; off <<= 1) mx = fmaxf(mx, __shfl_xor(mx, off, 32));
      const float mnew = fmaxf(mrow[r], mx);
      const float p0 = __expf(s[0][r] - mnew);
      const float p1 = __expf(s[1][r] - mnew);
      s[0][r] = p0; s[1][r] = p1;
      float rs = p0 + p1;
#pragma unroll
      for (int off = 1; off < 16; off <<= 1) rs += __shfl_xor(rs, off, 32);
      const float alpha = __expf(mrow[r] - mnew);
      lsum[r] = lsum[r] * alpha + rs;
      mrow[r] = mnew;
#pragma unroll
      for (int c = 0; c < 4; ++c) o[c][r] *= alpha;
    }

    // P (C layout) -> LDS -> A layout. Intra-wave LDS ops are in-order (DScnt).
#pragma unroll
    for (int st = 0; st < 2; ++st)
#pragma unroll
      for (int r = 0; r < 8; ++r)
        Pl[w][(half * 8 + r) * 32 + st * 16 + lrow] = (bf16)s[st][r];

    AV pa;
    pa.s.lo = *(const bf16x8*)&Pl[w][lrow * 32 + half * 16];
    pa.s.hi = *(const bf16x8*)&Pl[w][lrow * 32 + half * 16 + 8];

    // O += P V: B tile from LDS in B-layout (col = d, K = key = half*16 + e).
#pragma unroll
    for (int c = 0; c < 4; ++c) {
      union { bf16x16 v; bf16 e[16]; } bv;
#pragma unroll
      for (int e = 0; e < 16; ++e)
        bv.e[e] = Vl[w][(half * 16 + e) * 64 + c * 16 + lrow];
      o[c] = wmma_bf16(pa.v, bv.v, o[c]);
    }
  }

  // Normalize and store bf16, heads concatenated: [b, t, h*64 + d].
#pragma unroll
  for (int c = 0; c < 4; ++c)
#pragma unroll
    for (int r = 0; r < 8; ++r) {
      const float vv = o[c][r] / lsum[r];
      Ob[headOff + (size_t)(qBase + half * 8 + r) * E_ + c * 16 + lrow] = (bf16)vv;
    }
}

// ---------------------------------------------------------------------------
// Host-side orchestration
// ---------------------------------------------------------------------------
static inline size_t align256(size_t x) { return (x + 255) & ~(size_t)255; }

static void launch_gemm(const bf16* A, const bf16* W, const float* bias,
                        const float* resid, float* Cf, bf16* Cb,
                        int M, int N, int K, int relu, hipStream_t stream) {
  const int waves = (M / 32) * (N / 64);
  const int blocks = (waves + 3) / 4;
  gemm_bf16_wmma<<<blocks, 128, 0, stream>>>(A, W, bias, resid, Cf, Cb, M, N, K, relu);
}

extern "C" void kernel_launch(void* const* d_in, const int* in_sizes, int n_in,
                              void* d_out, int out_size, void* d_ws, size_t ws_size,
                              hipStream_t stream) {
  (void)in_sizes; (void)n_in; (void)out_size; (void)ws_size;

  const float* x   = (const float*)d_in[0];
  const float* Wq  = (const float*)d_in[1];
  const float* Wk  = (const float*)d_in[2];
  const float* Wv  = (const float*)d_in[3];
  const float* Wo  = (const float*)d_in[4];
  const float* bo  = (const float*)d_in[5];
  const float* W1  = (const float*)d_in[6];
  const float* b1  = (const float*)d_in[7];
  const float* W2  = (const float*)d_in[8];
  const float* b2  = (const float*)d_in[9];
  const float* g1  = (const float*)d_in[10];
  const float* be1 = (const float*)d_in[11];
  const float* g2  = (const float*)d_in[12];
  const float* be2 = (const float*)d_in[13];
  float* out = (float*)d_out;

  // Workspace partition (~216 MB total)
  char* ws = (char*)d_ws;
  size_t off = 0;
  auto take = [&](size_t bytes) { char* p = ws + off; off += align256(bytes); return p; };

  bf16* WqB = (bf16*)take((size_t)E_ * E_ * 2);
  bf16* WkB = (bf16*)take((size_t)E_ * E_ * 2);
  bf16* WvB = (bf16*)take((size_t)E_ * E_ * 2);
  bf16* WoB = (bf16*)take((size_t)E_ * E_ * 2);
  bf16* W1B = (bf16*)take((size_t)FF_ * E_ * 2);
  bf16* W2B = (bf16*)take((size_t)E_ * FF_ * 2);
  bf16* h1  = (bf16*)take((size_t)BT_ * E_ * 2);
  bf16* Qb  = (bf16*)take((size_t)BT_ * E_ * 2);
  bf16* Kb  = (bf16*)take((size_t)BT_ * E_ * 2);
  bf16* Vb  = (bf16*)take((size_t)BT_ * E_ * 2);
  bf16* Ab  = (bf16*)take((size_t)BT_ * E_ * 2);   // attention output, heads concat
  float* x1 = (float*)take((size_t)BT_ * E_ * 4);  // x + attn proj
  bf16* h2  = (bf16*)take((size_t)BT_ * E_ * 2);
  bf16* f1  = (bf16*)take((size_t)BT_ * FF_ * 2);  // relu(h2 W1^T + b1)

  // 1) Weight conversion fp32 -> bf16
  cvt_bf16_kernel<<<512, 256, 0, stream>>>(Wq, WqB, E_ * E_);
  cvt_bf16_kernel<<<512, 256, 0, stream>>>(Wk, WkB, E_ * E_);
  cvt_bf16_kernel<<<512, 256, 0, stream>>>(Wv, WvB, E_ * E_);
  cvt_bf16_kernel<<<512, 256, 0, stream>>>(Wo, WoB, E_ * E_);
  cvt_bf16_kernel<<<512, 256, 0, stream>>>(W1, W1B, FF_ * E_);
  cvt_bf16_kernel<<<512, 256, 0, stream>>>(W2, W2B, E_ * FF_);

  // 2) LN1
  ln_kernel<<<BT_, 256, 0, stream>>>(x, g1, be1, h1);

  // 3) Q/K/V projections (heads concat along columns, torch Linear layout)
  launch_gemm(h1, WqB, nullptr, nullptr, nullptr, Qb, BT_, E_, E_, 0, stream);
  launch_gemm(h1, WkB, nullptr, nullptr, nullptr, Kb, BT_, E_, E_, 0, stream);
  launch_gemm(h1, WvB, nullptr, nullptr, nullptr, Vb, BT_, E_, E_, 0, stream);

  // 4) Causal attention
  attn_kernel<<<(B_ * H_ * (T_ / 16)) / 4, 128, 0, stream>>>(Qb, Kb, Vb, Ab);

  // 5) Output projection + bo + residual(x) -> x1 (fp32)
  launch_gemm(Ab, WoB, bo, x, x1, nullptr, BT_, E_, E_, 0, stream);

  // 6) LN2
  ln_kernel<<<BT_, 256, 0, stream>>>(x1, g2, be2, h2);

  // 7) FFN up + ReLU (bf16 for the next GEMM)
  launch_gemm(h2, W1B, b1, nullptr, nullptr, f1, BT_, FF_, E_, 1, stream);

  // 8) FFN down + b2 + residual(x1) -> final output (fp32)
  launch_gemm(f1, W2B, b2, x1, out, nullptr, BT_, E_, FF_, 0, stream);
}
